// Attend_50036368998872
// MI455X (gfx1250) — matrix operand order repaired and checked by reading, not compiled, not run on hardware
//
#include <hip/hip_runtime.h>
#include <math.h>

// ---------------------------------------------------------------------------
// Problem constants (from the reference)
// ---------------------------------------------------------------------------
#define BH     16        // b*h = 2*8
#define HEADS  8
#define NSEQ   8192
#define DHEAD  64
#define SEGL   1024
#define NSEG   8         // NSEQ / SEGL
#define NQB    16        // q blocks of 64 rows per segment
#define LDP    72        // padded f16 LDS row stride (144 B, 16B-aligned)
#define SSTR   66        // padded f32 score LDS stride

typedef __attribute__((ext_vector_type(16))) _Float16 v16h;
typedef __attribute__((ext_vector_type(8)))  _Float16 v8h;
typedef __attribute__((ext_vector_type(8)))  float    v8f;

// ---------------------------------------------------------------------------
// WMMA fragment loaders, contiguous-K fast path (2 x ds_load_b128 each).
// A (16x32 f16): lane L: M=L&15, kb=(L<16)?0:8;  K runs [kb..kb+7],[16+kb..+7]
// B (32x16 f16): lane L: N=L&15, kb=(L<16)?0:8;  K runs [kb..kb+7],[16+kb..+7]
// C/D (16x16 f32): lane L: col=L&15, VGPR r: row = r + (L<16?0:8)
// ---------------------------------------------------------------------------
static __device__ __forceinline__ v16h frag_a_c(const _Float16* base, int mStride,
                                                int lane) {
  const int M  = lane & 15;
  const int kb = (lane < 16) ? 0 : 8;
  const _Float16* p = base + M * mStride + kb;
  v8h lo = *(const v8h*)(p);
  v8h hi = *(const v8h*)(p + 16);
  return __builtin_shufflevector(lo, hi, 0, 1, 2, 3, 4, 5, 6, 7,
                                 8, 9, 10, 11, 12, 13, 14, 15);
}

static __device__ __forceinline__ v16h frag_b_c(const _Float16* base, int nStride,
                                                int lane) {
  const int N  = lane & 15;
  const int kb = (lane < 16) ? 0 : 8;
  const _Float16* p = base + N * nStride + kb;
  v8h lo = *(const v8h*)(p);
  v8h hi = *(const v8h*)(p + 16);
  return __builtin_shufflevector(lo, hi, 0, 1, 2, 3, 4, 5, 6, 7,
                                 8, 9, 10, 11, 12, 13, 14, 15);
}

static __device__ __forceinline__ v8f wmma_f16(v16h a, v16h b, v8f c) {
  return __builtin_amdgcn_wmma_f32_16x16x32_f16(false, a, false, b,
                                                (short)0, c, false, false);
}

// Async global->LDS copy of 16 bytes per lane (gfx1250, ASYNCcnt-tracked).
// LDS offset = low 32 bits of the generic pointer (LDS aperture, ISA 10.2).
static __device__ __forceinline__ void async_copy_b128(void* lds_dst,
                                                       const void* gsrc) {
  const unsigned ldsoff = (unsigned)(unsigned long long)lds_dst;
  asm volatile("global_load_async_to_lds_b128 %0, %1, off"
               :
               : "v"(ldsoff), "v"(gsrc)
               : "memory");
}

static __device__ __forceinline__ void async_wait0() {
  asm volatile("s_wait_asynccnt 0x0" ::: "memory");
}

// ---------------------------------------------------------------------------
// Kernel 1: RoPE on q,k + fp16 packing of q,k,v
// ---------------------------------------------------------------------------
__global__ void rope_pack(const float* __restrict__ q, const float* __restrict__ k,
                          const float* __restrict__ v,
                          _Float16* __restrict__ qrot, _Float16* __restrict__ krot,
                          _Float16* __restrict__ vh) {
  const long gid = (long)blockIdx.x * blockDim.x + threadIdx.x;
  const long row = gid >> 5;            // bh*NSEQ + position
  if (row >= (long)BH * NSEQ) return;
  const int  j   = (int)(gid & 31);
  const int  pos = (int)(row & (NSEQ - 1));

  const float inv = __expf(-(float)j * (0.03125f * logf(10000.0f)));
  float sn, cs;
  __sincosf((float)pos * inv, &sn, &cs);

  const long i0 = row * DHEAD + j;
  const long i1 = i0 + 32;

  const float q1 = q[i0], q2 = q[i1];
  qrot[i0] = (_Float16)(q1 * cs - q2 * sn);
  qrot[i1] = (_Float16)(q2 * cs + q1 * sn);

  const float k1 = k[i0], k2 = k[i1];
  krot[i0] = (_Float16)(k1 * cs - k2 * sn);
  krot[i1] = (_Float16)(k2 * cs + k1 * sn);

  vh[i0] = (_Float16)v[i0];
  vh[i1] = (_Float16)v[i1];
}

// ---------------------------------------------------------------------------
// Kernel 2: per-segment  dM = (elu(k)+1)^T @ v  and  dz = sum(elu(k)+1)
// ke and v staged TRANSPOSED in LDS so WMMA K (= sequence rows) is contiguous.
// ---------------------------------------------------------------------------
__global__ void delta_kernel(const _Float16* __restrict__ krot,
                             const _Float16* __restrict__ vh,
                             float* __restrict__ dM, float* __restrict__ dz) {
  const int blk = blockIdx.x;
  const int bh  = blk >> 3;
  const int s   = blk & 7;

  __shared__ alignas(16) _Float16 Ket[64 * LDP];  // [d][nrow]
  __shared__ alignas(16) _Float16 Vt[64 * LDP];   // [e][nrow]
  __shared__ float zred[256];

  const int t    = threadIdx.x;
  const int w    = t >> 5;
  const int lane = t & 31;

  const v8f zero = {};
  v8f acc[2] = {zero, zero};
  float zacc = 0.0f;

  const long base = ((long)bh * NSEQ + (long)s * SEGL) * DHEAD;

  for (int ch = 0; ch < SEGL / 64; ++ch) {
    __syncthreads();
    const long cb = base + (long)ch * 64 * DHEAD;
    for (int i = t; i < 512; i += 256) {
      const int r = i >> 3;              // source row (sequence)
      const int c = (i & 7) * 8;         // source col (feature)
      v8h kv8 = *(const v8h*)(krot + cb + r * DHEAD + c);
      v8h vv8 = *(const v8h*)(vh   + cb + r * DHEAD + c);
#pragma unroll
      for (int jj = 0; jj < 8; ++jj) {
        const float kvf = (float)kv8[jj];
        const float ke  = (kvf > 0.0f) ? (kvf + 1.0f) : __expf(kvf);  // elu+1
        Ket[(c + jj) * LDP + r] = (_Float16)ke;
        Vt[(c + jj) * LDP + r]  = vv8[jj];
      }
      if (ch + 1 < SEGL / 64) {  // prefetch next chunk (global_prefetch_b8)
        __builtin_prefetch(krot + cb + 64 * DHEAD + r * DHEAD + c, 0, 3);
        __builtin_prefetch(vh   + cb + 64 * DHEAD + r * DHEAD + c, 0, 3);
      }
    }
    __syncthreads();

    {  // z partials: thread t sums 16 contiguous rows for feature d = t&63
      const int d = t & 63, grp = t >> 6;
      v8h a0 = *(const v8h*)(&Ket[d * LDP + grp * 16]);
      v8h a1 = *(const v8h*)(&Ket[d * LDP + grp * 16 + 8]);
#pragma unroll
      for (int r = 0; r < 8; ++r) zacc += (float)a0[r] + (float)a1[r];
    }

#pragma unroll
    for (int ti = 0; ti < 2; ++ti) {
      const int tt = w + ti * 8;
      const int dr = tt >> 2, ec = tt & 3;
#pragma unroll
      for (int k0 = 0; k0 < 64; k0 += 32) {
        v16h A = frag_a_c(Ket + dr * 16 * LDP + k0, LDP, lane);  // ke^T[M=d][K=n]
        v16h B = frag_b_c(Vt + ec * 16 * LDP + k0, LDP, lane);   // v[K=n][N=e]
        acc[ti] = wmma_f16(A, B, acc[ti]);
      }
    }
  }
  __syncthreads();

#pragma unroll
  for (int ti = 0; ti < 2; ++ti) {
    const int tt = w + ti * 8;
    const int dr = tt >> 2, ec = tt & 3;
#pragma unroll
    for (int r = 0; r < 8; ++r) {
      const int row = dr * 16 + r + ((lane >= 16) ? 8 : 0);
      const int col = ec * 16 + (lane & 15);
      dM[((long)bh * NSEG + s) * 4096 + row * 64 + col] = acc[ti][r];
    }
  }

  zred[t] = zacc;
  __syncthreads();
  if (t < 64)
    dz[((long)bh * NSEG + s) * 64 + t] =
        zred[t] + zred[t + 64] + zred[t + 128] + zred[t + 192];
}

// ---------------------------------------------------------------------------
// Kernel 3: exclusive prefix sums of dM/dz over segments (per bh)
// ---------------------------------------------------------------------------
__global__ void prefix_kernel(const float* __restrict__ dM, const float* __restrict__ dz,
                              float* __restrict__ Mpref, float* __restrict__ zpref) {
  const int bh = blockIdx.x;
  const int t  = threadIdx.x;
  for (int i = t; i < 4096 + 64; i += 256) {
    float run = 0.0f;
    if (i < 4096) {
      for (int s = 0; s < NSEG; ++s) {
        const long idx = ((long)bh * NSEG + s) * 4096 + i;
        Mpref[idx] = run;
        run += dM[idx];
      }
    } else {
      const int d = i - 4096;
      for (int s = 0; s < NSEG; ++s) {
        const long idx = ((long)bh * NSEG + s) * 64 + d;
        zpref[idx] = run;
        run += dz[idx];
      }
    }
  }
}

// ---------------------------------------------------------------------------
// Kernel 4: flash attention (causal, per segment) + linear-memory term + gate
// one block per (bh, seg, 64-row q block); 256 threads = 8 waves.
// K tile staged via async global->LDS (ASYNCcnt); V transposed on VGPR path.
// ---------------------------------------------------------------------------
__global__ void attn_kernel(const _Float16* __restrict__ qrot,
                            const _Float16* __restrict__ krot,
                            const _Float16* __restrict__ vh,
                            const float* __restrict__ Mpref,
                            const float* __restrict__ zpref,
                            const float* __restrict__ gate,
                            float* __restrict__ out) {
  const int blk = blockIdx.x;
  const int qb  = blk & (NQB - 1);
  const int s   = (blk >> 4) & (NSEG - 1);
  const int bh  = blk >> 7;
  const int h   = bh & (HEADS - 1);

  __shared__ alignas(16) _Float16 Qs[64 * LDP];   // [qrow][d]
  __shared__ alignas(16) _Float16 Ks[64 * LDP];   // [key][d]   (also M^T [e][d])
  __shared__ alignas(16) _Float16 Vt[64 * LDP];   // [d][key]   (transposed)
  __shared__ alignas(16) _Float16 Ps[64 * LDP];   // [qrow][key] (also qe [qrow][d])
  __shared__ float    Ssc[64 * SSTR];
  __shared__ float    m_i[64], l_i[64], alpha_s[64], den_s[64], zsh[64];
  __shared__ float    red[64 * 4];

  const int t     = threadIdx.x;
  const int w     = t >> 5;
  const int lane  = t & 31;
  const int rhalf = (lane >= 16) ? 8 : 0;

  const float g     = 1.0f / (1.0f + __expf(-gate[h]));
  const float scale = 0.125f;  // 1/sqrt(64)

  const long qbase = ((long)bh * NSEQ + (long)s * SEGL + (long)qb * 64) * DHEAD;

  // load Q tile (b128 both sides), init row stats
  for (int i = t; i < 512; i += 256) {
    const int r = i >> 3, c = (i & 7) * 8;
    *(v8h*)(&Qs[r * LDP + c]) = *(const v8h*)(qrot + qbase + r * DHEAD + c);
  }
  if (t < 64) { m_i[t] = -1e30f; l_i[t] = 0.0f; }
  __syncthreads();

  // hoist Q A-fragments (invariant over the flash loop): [tile][k0]
  v16h qa[2][2];
#pragma unroll
  for (int ti = 0; ti < 2; ++ti) {
    const int qr = (w + ti * 8) >> 2;
#pragma unroll
    for (int kk = 0; kk < 2; ++kk)
      qa[ti][kk] = frag_a_c(Qs + qr * 16 * LDP + kk * 32, LDP, lane);
  }

  const v8f zero = {};
  v8f o[2] = {zero, zero};

  // ------------------- flash loop over causal key chunks -------------------
  for (int kc = 0; kc <= qb; ++kc) {
    __syncthreads();
    const long kb = ((long)bh * NSEQ + (long)s * SEGL + (long)kc * 64) * DHEAD;
    for (int i = t; i < 512; i += 256) {
      const int r = i >> 3, c = (i & 7) * 8;
      // K row-major: async global->LDS b128, no VGPR round-trip
      async_copy_b128(&Ks[r * LDP + c], krot + kb + r * DHEAD + c);
      // V transposed: VGPR path (b128 global load + strided b16 LDS stores)
      v8h vv8 = *(const v8h*)(vh + kb + r * DHEAD + c);
#pragma unroll
      for (int jj = 0; jj < 8; ++jj) Vt[(c + jj) * LDP + r] = vv8[jj];
      if (kc < qb) {  // prefetch next chunk (global_prefetch_b8)
        __builtin_prefetch(krot + kb + 64 * DHEAD + r * DHEAD + c, 0, 3);
        __builtin_prefetch(vh   + kb + 64 * DHEAD + r * DHEAD + c, 0, 3);
      }
    }
    async_wait0();
    __syncthreads();

    // S = Q @ K^T * scale
#pragma unroll
    for (int ti = 0; ti < 2; ++ti) {
      const int tt = w + ti * 8;
      const int qr = tt >> 2, nc = tt & 3;
      v8f c = {};
#pragma unroll
      for (int kk = 0; kk < 2; ++kk) {
        v16h B = frag_b_c(Ks + nc * 16 * LDP + kk * 32, LDP, lane);  // K^T[K=d][N=key]
        c = wmma_f16(qa[ti][kk], B, c);
      }
#pragma unroll
      for (int r = 0; r < 8; ++r) {
        const int row = qr * 16 + r + rhalf;
        const int col = nc * 16 + (lane & 15);
        Ssc[row * SSTR + col] = c[r] * scale;
      }
    }
    __syncthreads();

    // online softmax: row max pass
    const int rr = t >> 2, qq = t & 3;
    const int row_g = qb * 64 + rr;
    {
      float mx = -1e30f;
#pragma unroll
      for (int c16 = 0; c16 < 16; ++c16) {
        const int col = qq * 16 + c16;
        const bool masked = (kc * 64 + col) > row_g;
        const float x = masked ? -1e30f : Ssc[rr * SSTR + col];
        mx = fmaxf(mx, x);
      }
      red[rr * 4 + qq] = mx;
    }
    __syncthreads();
    if (t < 64) {
      const float mx = fmaxf(fmaxf(red[t * 4], red[t * 4 + 1]),
                             fmaxf(red[t * 4 + 2], red[t * 4 + 3]));
      const float nm = fmaxf(m_i[t], mx);
      alpha_s[t] = __expf(m_i[t] - nm);
      m_i[t] = nm;
    }
    __syncthreads();

    // p = exp(s - m) pass, row sums
    {
      float ps = 0.0f;
#pragma unroll
      for (int c16 = 0; c16 < 16; ++c16) {
        const int col = qq * 16 + c16;
        const bool masked = (kc * 64 + col) > row_g;
        const float p = masked ? 0.0f : __expf(Ssc[rr * SSTR + col] - m_i[rr]);
        Ps[rr * LDP + col] = (_Float16)p;
        ps += p;
      }
      red[rr * 4 + qq] = ps;
    }
    __syncthreads();
    if (t < 64)
      l_i[t] = l_i[t] * alpha_s[t] +
               (red[t * 4] + red[t * 4 + 1] + red[t * 4 + 2] + red[t * 4 + 3]);
    __syncthreads();

    // O = O*alpha + P @ V
#pragma unroll
    for (int ti = 0; ti < 2; ++ti) {
      const int tt = w + ti * 8;
      const int qr = tt >> 2, dc = tt & 3;
#pragma unroll
      for (int r = 0; r < 8; ++r)
        o[ti][r] *= alpha_s[qr * 16 + r + rhalf];
#pragma unroll
      for (int kk = 0; kk < 2; ++kk) {
        v16h A = frag_a_c(Ps + qr * 16 * LDP + kk * 32, LDP, lane);  // P[M][K=key]
        v16h B = frag_b_c(Vt + dc * 16 * LDP + kk * 32, LDP, lane);  // V[K=key][N=d]
        o[ti] = wmma_f16(A, B, o[ti]);
      }
    }
  }

  // ------------------- linear-memory term: num = qe @ Mpref ----------------
  __syncthreads();
  const long mb = ((long)bh * NSEG + s) * 4096;
  const long zb = ((long)bh * NSEG + s) * 64;
  for (int i = t; i < 512; i += 256) {
    const int r = i >> 3, c = (i & 7) * 8;
    v8h q8 = *(const v8h*)(&Qs[r * LDP + c]);
    v8h e8;
#pragma unroll
    for (int jj = 0; jj < 8; ++jj) {
      const float qv = (float)q8[jj];
      e8[jj] = (_Float16)((qv > 0.0f) ? (qv + 1.0f) : __expf(qv));  // elu+1
      // Mpref transposed into Ks: Ks[e][d] = M[d][e]
      Ks[(c + jj) * LDP + r] = (_Float16)Mpref[mb + r * DHEAD + c + jj];
    }
    *(v8h*)(&Ps[r * LDP + c]) = e8;
  }
  if (t < 64) zsh[t] = zpref[zb + t];
  __syncthreads();

  {  // den = qe . z
    const int rr = t >> 2, qq = t & 3;
    float p = 0.0f;
#pragma unroll
    for (int c16 = 0; c16 < 16; ++c16) {
      const int col = qq * 16 + c16;
      p += (float)Ps[rr * LDP + col] * zsh[col];
    }
    red[rr * 4 + qq] = p;
  }
  __syncthreads();
  if (t < 64)
    den_s[t] = red[t * 4] + red[t * 4 + 1] + red[t * 4 + 2] + red[t * 4 + 3] + 1e-6f;
  __syncthreads();

  // NUM tiles + combine + store
#pragma unroll
  for (int ti = 0; ti < 2; ++ti) {
    const int tt = w + ti * 8;
    const int qr = tt >> 2, dc = tt & 3;
    v8f nmm = {};
#pragma unroll
    for (int kk = 0; kk < 2; ++kk) {
      v16h A = frag_a_c(Ps + qr * 16 * LDP + kk * 32, LDP, lane);   // qe[M][K=d]
      v16h B = frag_b_c(Ks + dc * 16 * LDP + kk * 32, LDP, lane);   // M^T[K=d][N=e]
      nmm = wmma_f16(A, B, nmm);
    }
#pragma unroll
    for (int r = 0; r < 8; ++r) {
      const int row = qr * 16 + r + rhalf;
      const int col = dc * 16 + (lane & 15);
      const float attn = o[ti][r] / (l_i[row] + 1e-30f);
      const float memo = nmm[r] / den_s[row];
      out[qbase + row * 64 + col] = g * memo + (1.0f - g) * attn;
    }
  }
}

// ---------------------------------------------------------------------------
// Host-side launcher
// ---------------------------------------------------------------------------
extern "C" void kernel_launch(void* const* d_in, const int* in_sizes, int n_in,
                              void* d_out, int out_size, void* d_ws, size_t ws_size,
                              hipStream_t stream) {
  const float* q    = (const float*)d_in[0];
  const float* k    = (const float*)d_in[1];
  const float* v    = (const float*)d_in[2];
  const float* gate = (const float*)d_in[3];
  float* out = (float*)d_out;

  const size_t NE = (size_t)BH * NSEQ * DHEAD;  // 8,388,608 elements

  char* ws = (char*)d_ws;
  _Float16* qrot  = (_Float16*)(ws);                       // NE * 2
  _Float16* krot  = (_Float16*)(ws + NE * 2);              // NE * 2
  _Float16* vh    = (_Float16*)(ws + NE * 4);              // NE * 2
  float*    dM    = (float*)(ws + NE * 6);                 // BH*NSEG*4096 * 4
  float*    dz    = (float*)(ws + NE * 6 + (size_t)BH * NSEG * 4096 * 4);
  float*    Mpref = (float*)(ws + NE * 6 + (size_t)BH * NSEG * (4096 + 64) * 4);
  float*    zpref = (float*)(ws + NE * 6 + (size_t)BH * NSEG * (4096 + 64 + 4096) * 4);

  {
    const long total = (long)BH * NSEQ * 32;
    rope_pack<<<(unsigned)(total / 256), 256, 0, stream>>>(q, k, v, qrot, krot, vh);
  }
  delta_kernel<<<BH * NSEG, 256, 0, stream>>>(krot, vh, dM, dz);
  prefix_kernel<<<BH, 256, 0, stream>>>(dM, dz, Mpref, zpref);
  attn_kernel<<<BH * NSEG * NQB, 256, 0, stream>>>(qrot, krot, vh, Mpref, zpref,
                                                   gate, out);
}
